// MultiHeadAttention_13391708029856
// MI455X (gfx1250) — matrix operand order repaired and checked by reading, not compiled, not run on hardware
//
#include <hip/hip_runtime.h>
#include <hip/hip_bf16.h>
#include <stdint.h>

// ---------------------------------------------------------------------------
// MultiHeadAttention forward for MI455X (gfx1250), bf16 WMMA path.
//   x:[2,2048,1024] f32 -> out:[2,2048,1024] f32
// ---------------------------------------------------------------------------

typedef __attribute__((ext_vector_type(16))) __bf16 v16bf;
typedef __attribute__((ext_vector_type(8)))  __bf16 v8bf;
typedef __attribute__((ext_vector_type(8)))  float  v8f;

#define DM   1024
#define SEQ  2048
#define NB   2
#define NH   16
#define DH   64
#define MTOT (NB * SEQ)   // 4096 rows for the projection GEMMs

static __device__ __forceinline__ unsigned short f2bf_bits(float f) {
  union { float f; unsigned u; } v; v.f = f;
  unsigned r = v.u + 0x7FFFu + ((v.u >> 16) & 1u);   // round-to-nearest-even
  return (unsigned short)(r >> 16);
}

static __device__ __forceinline__ v16bf comb16(v8bf lo, v8bf hi) {
  v16bf r;
#pragma unroll
  for (int i = 0; i < 8; i++) { r[i] = lo[i]; r[i + 8] = hi[i]; }
  return r;
}

// LDS byte offset of a __shared__ object (generic -> addrspace(3) -> int)
static __device__ __forceinline__ unsigned lds_addr(const void* p) {
  return (unsigned)(unsigned long long)(__attribute__((address_space(3))) const void*)p;
}

// async 16B global -> LDS copy (per active lane), tracked by ASYNCcnt
static __device__ __forceinline__ void async_cp16(unsigned lds_off,
                                                  const unsigned short* g) {
  asm volatile("global_load_async_to_lds_b128 %0, %1, off"
               :: "v"(lds_off), "v"(g) : "memory");
}

// reductions within a 16-lane half of the wave (rows of a 16x16 f32 C tile)
static __device__ __forceinline__ float red16_max(float v) {
#pragma unroll
  for (int m = 1; m < 16; m <<= 1) v = fmaxf(v, __shfl_xor(v, m, 32));
  return v;
}
static __device__ __forceinline__ float red16_sum(float v) {
#pragma unroll
  for (int m = 1; m < 16; m <<= 1) v += __shfl_xor(v, m, 32);
  return v;
}

// ---------------------------------------------------------------------------
// f32 -> bf16 conversion (grid-stride)
// ---------------------------------------------------------------------------
__global__ void cvt_f32_bf16(const float* __restrict__ src,
                             unsigned short* __restrict__ dst, int n) {
  int i = blockIdx.x * blockDim.x + threadIdx.x;
  int stride = gridDim.x * blockDim.x;
  for (; i < n; i += stride) dst[i] = f2bf_bits(src[i]);
}

// ---------------------------------------------------------------------------
// GEMM: out[m,n] = (A[m,:] . W[n,:]) * scale + bias[n]*scale
//   A : [M=4096, 1024] bf16 row-major;  W : [1024,1024] bf16 [out,in]
// MODE 0: bf16 out [b][h][s][d]   (Q, K)
// MODE 1: bf16 out [b][h][d][s]   (V transposed for PV B-fragments)
// MODE 2: f32  out [m][n]         (final projection)
// Block = 256 thr (8 waves), block tile 128(M) x 64(N); the shared W tile
// (64 x 64 k-slab, 8 KB) is staged via async global->LDS copies, double
// buffered; each wave then runs 8 x v_wmma_f32_16x16x32_bf16 per slab.
// ---------------------------------------------------------------------------
template <int MODE>
__global__ void __launch_bounds__(256)
gemm_bf16_wmma(const unsigned short* __restrict__ A,
               const unsigned short* __restrict__ W,
               const float* __restrict__ bias,
               unsigned short* __restrict__ outb,
               float* __restrict__ outf,
               float scale) {
  __shared__ __align__(16) unsigned short bsh[2][64 * 64];  // 2 x 8 KB

  const int tid  = threadIdx.x;
  const int lane = tid & 31;
  const int wv   = tid >> 5;
  const int l16  = lane & 15;
  const int half = lane >> 4;

  const int m0 = blockIdx.y * 128 + wv * 16;   // wave's M tile
  const int n0 = blockIdx.x * 64;              // block's N range

  // Stage one 64(N) x 64(K) bf16 W-slab into LDS buffer `buf`:
  // 512 16-byte chunks, 2 per thread, layout [n_local][k_local] row-major.
  auto stage = [&](int buf, int kbase) {
#pragma unroll
    for (int j = 0; j < 2; j++) {
      const int chunk = tid + j * 256;           // 0..511
      const int row = chunk >> 3, sub = chunk & 7;
      const unsigned short* g = W + (size_t)(n0 + row) * DM + kbase + sub * 8;
      async_cp16(lds_addr(&bsh[buf][(size_t)chunk * 8]), g);
    }
  };

  // A-fragment (16x32 bf16): lane row = lane%16; K chunk kb..kb+15,
  // kb = (lane/16)*16 -> two contiguous 16B loads per k-step.
  const unsigned short* Ap = A + (size_t)(m0 + l16) * DM + half * 16;

  v8f acc[4];
#pragma unroll
  for (int t = 0; t < 4; t++)
#pragma unroll
    for (int i = 0; i < 8; i++) acc[t][i] = 0.0f;

  stage(0, 0);
  int buf = 0;

#pragma unroll 1
  for (int k = 0; k < DM; k += 64) {
    if (k + 64 < DM) {
      stage(buf ^ 1, k + 64);
      asm volatile("s_wait_asynccnt 0x2" ::: "memory");  // prev slab landed
    } else {
      asm volatile("s_wait_asynccnt 0x0" ::: "memory");
    }
    __syncthreads();   // slab visible to all 8 waves

#pragma unroll
    for (int ks = 0; ks < 2; ks++) {
      const int kk = k + ks * 32;
      v16bf a = comb16(*(const v8bf*)(Ap + kk),
                       *(const v8bf*)(Ap + kk + 8));
      __builtin_prefetch((const void*)(Ap + kk + 256), 0, 3);
#pragma unroll
      for (int t = 0; t < 4; t++) {
        // B-fragment (32x16): lane col = t*16+l16, K = ks*32 + half*16 + 0..15
        const unsigned short* bp =
            &bsh[buf][(size_t)(t * 16 + l16) * 64 + ks * 32 + half * 16];
        v16bf b = *(const v16bf*)bp;
        acc[t] = __builtin_amdgcn_wmma_f32_16x16x32_bf16(
            false, a, false, b, (short)0, acc[t], false, false);
      }
    }
    __syncthreads();   // all reads done before slab is overwritten
    buf ^= 1;
  }

  // Epilogue. C layout: lane col = l16 (+16*t), rows = half*8 + v across regs.
#pragma unroll
  for (int t = 0; t < 4; t++) {
    const int n = n0 + t * 16 + l16;
    const float bv = bias[n];
#pragma unroll
    for (int v = 0; v < 8; v++) {
      const int mrow = m0 + half * 8 + v;
      const float val = (acc[t][v] + bv) * scale;
      if (MODE == 0) {
        const int b = mrow >> 11, s = mrow & (SEQ - 1);
        const int h = n >> 6,    d = n & (DH - 1);
        outb[((size_t)(b * NH + h) * SEQ + s) * DH + d] = f2bf_bits(val);
      } else if (MODE == 1) {
        const int b = mrow >> 11, s = mrow & (SEQ - 1);
        const int h = n >> 6,    d = n & (DH - 1);
        outb[((size_t)(b * NH + h) * DH + d) * SEQ + s] = f2bf_bits(val);
      } else {
        outf[(size_t)mrow * DM + n] = val;
      }
    }
  }
}

// ---------------------------------------------------------------------------
// Flash-style attention. Q,K: [b][h][s][64] bf16 (Q pre-scaled by 1/8),
// Vt: [b][h][64][s] bf16. Out: [b][s][1024] bf16.
// Block = 128 threads (4 waves); wave owns 16 Q rows; KV chunked by 64.
// ---------------------------------------------------------------------------
__global__ void __launch_bounds__(128)
attn_fwd_wmma(const unsigned short* __restrict__ Q,
              const unsigned short* __restrict__ Kt,
              const unsigned short* __restrict__ Vt,
              unsigned short* __restrict__ Ob) {
  const int lane = threadIdx.x & 31;
  const int wv   = threadIdx.x >> 5;
  const int l16  = lane & 15;
  const int half = lane >> 4;

  const int b = blockIdx.z, h = blockIdx.y;
  const int qrow0 = blockIdx.x * 64 + wv * 16;

  const size_t bh = (size_t)(b * NH + h);
  const unsigned short* Qbh = Q  + bh * SEQ * DH;
  const unsigned short* Kbh = Kt + bh * SEQ * DH;
  const unsigned short* Vbh = Vt + bh * DH * SEQ;

  // LDS bounce buffer: P tile per wave, 16 rows x 64 cols bf16 (+pad).
  __shared__ __align__(16) unsigned short pb[4][16][72];

  // Load Q A-fragments for the two 32-wide K-steps of d_head=64.
  const unsigned short* qp = Qbh + (size_t)(qrow0 + l16) * DH + half * 16;
  v16bf qf[2];
#pragma unroll
  for (int ks = 0; ks < 2; ks++)
    qf[ks] = comb16(*(const v8bf*)(qp + ks * 32),
                    *(const v8bf*)(qp + ks * 32 + 8));

  v8f o[4];
  float m_[8], l_[8], alpha[8];
#pragma unroll
  for (int t = 0; t < 4; t++)
#pragma unroll
    for (int i = 0; i < 8; i++) o[t][i] = 0.0f;
#pragma unroll
  for (int v = 0; v < 8; v++) { m_[v] = -3.0e38f; l_[v] = 0.0f; }

#pragma unroll 1
  for (int kv0 = 0; kv0 < SEQ; kv0 += 64) {
    // ---- S = Q . K^T (16 x 64 tile) -------------------------------------
    v8f sacc[4];
#pragma unroll
    for (int t = 0; t < 4; t++)
#pragma unroll
      for (int i = 0; i < 8; i++) sacc[t][i] = 0.0f;

#pragma unroll
    for (int t = 0; t < 4; t++) {
      // B-fragment of K^T: lane col j = kv0+t*16+l16; K-dim = d, contiguous.
      const unsigned short* kp =
          Kbh + (size_t)(kv0 + t * 16 + l16) * DH + half * 16;
      __builtin_prefetch((const void*)(kp + 64 * DH), 0, 3);
#pragma unroll
      for (int ks = 0; ks < 2; ks++) {
        v16bf kf = *(const v16bf*)(kp + ks * 32);
        sacc[t] = __builtin_amdgcn_wmma_f32_16x16x32_bf16(
            false, qf[ks], false, kf, (short)0, sacc[t], false, false);
      }
    }

    // ---- online softmax (rows live in 16-lane halves) -------------------
#pragma unroll
    for (int v = 0; v < 8; v++) {
      float rm = fmaxf(fmaxf(sacc[0][v], sacc[1][v]),
                       fmaxf(sacc[2][v], sacc[3][v]));
      rm = red16_max(rm);
      const float mn = fmaxf(m_[v], rm);
      const float al = __expf(m_[v] - mn);
      float rs = 0.0f;
#pragma unroll
      for (int t = 0; t < 4; t++) {
        const float p = __expf(sacc[t][v] - mn);
        sacc[t][v] = p;
        rs += p;
      }
      rs = red16_sum(rs);
      l_[v] = l_[v] * al + rs;
      m_[v] = mn;
      alpha[v] = al;
    }

    // ---- P: C-layout -> A-layout via per-wave LDS bounce ----------------
#pragma unroll
    for (int t = 0; t < 4; t++)
#pragma unroll
      for (int v = 0; v < 8; v++)
        pb[wv][half * 8 + v][t * 16 + l16] = f2bf_bits(sacc[t][v]);

    asm volatile("s_wait_dscnt 0x0" ::: "memory");

    const unsigned short* prow = &pb[wv][l16][0];
    v16bf pa[2];
#pragma unroll
    for (int ks = 0; ks < 2; ks++)
      pa[ks] = comb16(*(const v8bf*)(prow + ks * 32 + half * 16),
                      *(const v8bf*)(prow + ks * 32 + half * 16 + 8));

    // ---- O = O*alpha + P . V --------------------------------------------
#pragma unroll
    for (int t = 0; t < 4; t++) {
#pragma unroll
      for (int v = 0; v < 8; v++) o[t][v] *= alpha[v];
#pragma unroll
      for (int ks = 0; ks < 2; ks++) {
        // B-fragment of V: lane col d = t*16+l16; K-dim = kv, contiguous in Vt.
        const unsigned short* vp =
            Vbh + (size_t)(t * 16 + l16) * SEQ + kv0 + ks * 32 + half * 16;
        v16bf vf = *(const v16bf*)vp;
        o[t] = __builtin_amdgcn_wmma_f32_16x16x32_bf16(
            false, pa[ks], false, vf, (short)0, o[t], false, false);
      }
    }
  }

  // ---- normalize and store to [b][s][d_model] bf16 ----------------------
  float linv[8];
#pragma unroll
  for (int v = 0; v < 8; v++) linv[v] = 1.0f / l_[v];

#pragma unroll
  for (int t = 0; t < 4; t++) {
    const int col = h * DH + t * 16 + l16;
#pragma unroll
    for (int v = 0; v < 8; v++) {
      const int srow = qrow0 + half * 8 + v;
      Ob[((size_t)b * SEQ + srow) * DM + col] = f2bf_bits(o[t][v] * linv[v]);
    }
  }
}

// ---------------------------------------------------------------------------
// Host launcher
// ---------------------------------------------------------------------------
extern "C" void kernel_launch(void* const* d_in, const int* in_sizes, int n_in,
                              void* d_out, int out_size, void* d_ws, size_t ws_size,
                              hipStream_t stream) {
  const float* x   = (const float*)d_in[0];
  const float* w_q = (const float*)d_in[1];
  const float* b_q = (const float*)d_in[2];
  const float* w_k = (const float*)d_in[3];
  const float* b_k = (const float*)d_in[4];
  const float* w_v = (const float*)d_in[5];
  const float* b_v = (const float*)d_in[6];
  const float* w_o = (const float*)d_in[7];
  const float* b_o = (const float*)d_in[8];
  float* out = (float*)d_out;

  char* ws = (char*)d_ws;
  const size_t MB = 1u << 20;
  unsigned short* xb  = (unsigned short*)(ws);             // 8 MB  x bf16
  unsigned short* wqb = (unsigned short*)(ws + 8  * MB);   // 2 MB
  unsigned short* wkb = (unsigned short*)(ws + 10 * MB);   // 2 MB
  unsigned short* wvb = (unsigned short*)(ws + 12 * MB);   // 2 MB
  unsigned short* wob = (unsigned short*)(ws + 14 * MB);   // 2 MB
  unsigned short* qb  = (unsigned short*)(ws + 16 * MB);   // 8 MB  Q [b][h][s][d]
  unsigned short* kb  = (unsigned short*)(ws + 24 * MB);   // 8 MB  K [b][h][s][d]
  unsigned short* vtb = (unsigned short*)(ws + 32 * MB);   // 8 MB  V [b][h][d][s]
  unsigned short* ab  = (unsigned short*)(ws + 40 * MB);   // 8 MB  attn out bf16

  // f32 -> bf16 conversions
  cvt_f32_bf16<<<1024, 256, 0, stream>>>(x,   xb,  MTOT * DM);
  cvt_f32_bf16<<<512,  256, 0, stream>>>(w_q, wqb, DM * DM);
  cvt_f32_bf16<<<512,  256, 0, stream>>>(w_k, wkb, DM * DM);
  cvt_f32_bf16<<<512,  256, 0, stream>>>(w_v, wvb, DM * DM);
  cvt_f32_bf16<<<512,  256, 0, stream>>>(w_o, wob, DM * DM);

  dim3 ggrid(DM / 64, MTOT / 128);   // (16, 32)
  // Q (scaled by 1/sqrt(64)=0.125), K, V(transposed layout)
  gemm_bf16_wmma<0><<<ggrid, 256, 0, stream>>>(xb, wqb, b_q, qb,  nullptr, 0.125f);
  gemm_bf16_wmma<0><<<ggrid, 256, 0, stream>>>(xb, wkb, b_k, kb,  nullptr, 1.0f);
  gemm_bf16_wmma<1><<<ggrid, 256, 0, stream>>>(xb, wvb, b_v, vtb, nullptr, 1.0f);

  // attention: grid (S/64, H, B), 128 threads (4 waves)
  attn_fwd_wmma<<<dim3(SEQ / 64, NH, NB), 128, 0, stream>>>(qb, kb, vtb, ab);

  // output projection -> f32 d_out
  gemm_bf16_wmma<2><<<ggrid, 256, 0, stream>>>(ab, wob, b_o, nullptr, out, 1.0f);
}
